// TransformerRegressor_2525440770253
// MI455X (gfx1250) — matrix operand order repaired and checked by reading, not compiled, not run on hardware
//
#include <hip/hip_runtime.h>
#include <hip/hip_bf16.h>
#include <math.h>

// ---------------- model constants ----------------
#define B_      8
#define SEQ_    10
#define EMBED_  64
#define HEADS_  4
#define MDIM_   512
#define DI_     1024
#define NST_    16
#define DTR_    32
#define EPS_    1e-6f

typedef __bf16 bf16_t;
typedef __attribute__((ext_vector_type(16))) __bf16 v16bf;
typedef __attribute__((ext_vector_type(8)))  float  v8f;

// ---------------- device math helpers ----------------
__device__ __forceinline__ float silu_f(float x) { return x / (1.f + __expf(-x)); }
__device__ __forceinline__ float gelu_f(float x) { return 0.5f * x * (1.f + erff(x * 0.70710678f)); }
__device__ __forceinline__ float softplus_f(float x) { return (x > 20.f) ? x : log1pf(__expf(x)); }
__device__ __forceinline__ float act_f(float v, int act) {
  if (act == 1) return silu_f(v);
  if (act == 2) return gelu_f(v);
  if (act == 3) return softplus_f(v);
  return v;
}

// LDS offset of a __shared__ object: low 32 bits of the generic address
// (ISA 10.2: LDS aperture maps addr[31:0] to the wave's LDS space).
__device__ __forceinline__ unsigned lds_off(const void* p) {
  return (unsigned)(unsigned long long)p;
}
// Async global->LDS 16-byte copy (GLOBAL_LOAD_ASYNC_TO_LDS_B128, ASYNCcnt).
__device__ __forceinline__ void async_ld_b128(unsigned ldsaddr, const void* gaddr) {
  asm volatile("global_load_async_to_lds_b128 %0, %1, off"
               :: "v"(ldsaddr), "v"((unsigned long long)gaddr) : "memory");
}
__device__ __forceinline__ void wait_async0() {
  asm volatile("s_wait_asynccnt 0x0" ::: "memory");
}

// Cubic B-spline basis on uniform grid: grid[i] = (i-3)*0.4 - 1, 12 knots -> 8 funcs
__device__ __forceinline__ void bsplines(float x, float* out) {
  const float h = 0.4f;
  float b[11];
#pragma unroll
  for (int j = 0; j < 11; ++j) {
    float gj  = (float)(j - 3) * h - 1.0f;
    float gj1 = (float)(j - 2) * h - 1.0f;
    b[j] = (x >= gj && x < gj1) ? 1.f : 0.f;
  }
#pragma unroll
  for (int i = 1; i <= 3; ++i) {
    float inv = 1.0f / (h * (float)i);
#pragma unroll
    for (int j = 0; j < 10; ++j) {
      if (j < 11 - i) {
        float gj    = (float)(j - 3) * h - 1.0f;
        float gjip1 = (float)(j + i - 2) * h - 1.0f;
        b[j] = (x - gj) * inv * b[j] + (gjip1 - x) * inv * b[j + 1];
      }
    }
  }
#pragma unroll
  for (int k = 0; k < 8; ++k) out[k] = b[k];
}

// ---------------- WMMA GEMM: C = act(A * B^T + bias) ----------------
// A: M x K bf16 (lda), B: N x K bf16 (ldb), C: M x N (f32 or bf16).
// Block = 128 threads (4 waves). Block tile: 16(M) x 64(N); K slab = 64.
// Aligned full tiles stream via GLOBAL_LOAD_ASYNC_TO_LDS_B128; edges use
// branchless clamped fills (no per-element exec masking).
#define KT_ 64
__global__ void gemm_bf16_wmma(const bf16_t* __restrict__ A, int lda,
                               const bf16_t* __restrict__ Bw, int ldb,
                               const float* __restrict__ bias,
                               void* __restrict__ Cout, int ldc, int out_bf,
                               int M, int N, int Kdim, int act)
{
  __shared__ bf16_t As[16 * KT_];
  __shared__ bf16_t Bs[4][16 * KT_];
  const int wave = threadIdx.x >> 5;
  const int lane = threadIdx.x & 31;
  const int m0 = blockIdx.y * 16;
  const int n0 = blockIdx.x * 64 + wave * 16;
  const bool aAligned = (lda & 7) == 0;   // 16B-aligned rows for b128 path
  const bool bAligned = (ldb & 7) == 0;
  v8f acc = {};

  for (int k0 = 0; k0 < Kdim; k0 += KT_) {
    const bool kfull = (k0 + KT_) <= Kdim;
    // ---- A tile 16 x 64 bf16 = 128 x 16B chunks, one chunk per thread ----
    {
      const int c = threadIdx.x;       // chunk id
      const int r = c >> 3;            // 8 chunks per row
      const int col = (c & 7) * 8;     // bf16 column
      if (kfull && aAligned && (m0 + 16) <= M) {
        async_ld_b128(lds_off(&As[r * KT_ + col]),
                      A + (size_t)(m0 + r) * lda + k0 + col);
      } else {
        const bool rok = (m0 + r) < M;
        const int rr = rok ? (m0 + r) : (M - 1);
#pragma unroll
        for (int e = 0; e < 8; ++e) {
          int kc = k0 + col + e;
          int kcl = (kc < Kdim) ? kc : (Kdim - 1);
          bf16_t v = A[(size_t)rr * lda + kcl];
          if (!rok || kc >= Kdim) v = (bf16_t)0.f;
          As[r * KT_ + col + e] = v;
        }
      }
    }
    // ---- B tile per wave: 16 x 64 bf16 = 128 chunks, 4 per lane ----
    {
      const bool bfull = kfull && bAligned && (n0 + 16) <= N;
#pragma unroll
      for (int cc = 0; cc < 4; ++cc) {
        const int c = cc * 32 + lane;
        const int r = c >> 3;
        const int col = (c & 7) * 8;
        if (bfull) {
          async_ld_b128(lds_off(&Bs[wave][r * KT_ + col]),
                        Bw + (size_t)(n0 + r) * ldb + k0 + col);
        } else {
          const bool rok = (n0 + r) < N;
          const int rr = rok ? (n0 + r) : (N - 1);
#pragma unroll
          for (int e = 0; e < 8; ++e) {
            int kc = k0 + col + e;
            int kcl = (kc < Kdim) ? kc : (Kdim - 1);
            bf16_t v = Bw[(size_t)rr * ldb + kcl];
            if (!rok || kc >= Kdim) v = (bf16_t)0.f;
            Bs[wave][r * KT_ + col + e] = v;
          }
        }
      }
    }
    if (lane == 0 && (k0 + KT_) < Kdim)
      __builtin_prefetch(&A[(size_t)m0 * lda + (k0 + KT_)], 0, 1);
    wait_async0();
    __syncthreads();

    // ---- two 16x16x32 WMMA steps over the 64-wide slab ----
    const int mrow = lane & 15;
    const int kbA  = (lane < 16) ? 0 : 8;
    const int ncol = lane & 15;
    const int kbB  = (lane < 16) ? 0 : 8;
#pragma unroll
    for (int kk = 0; kk < 2; ++kk) {
      v16bf af, bfb;
      const int abase = mrow * KT_ + kk * 32;
#pragma unroll
      for (int e = 0; e < 16; ++e) {
        int v = e >> 1, p = e & 1;
        int ko = ((v < 4) ? 0 : 16) + kbA + 2 * (v & 3) + p;
        af[e] = As[abase + ko];
      }
      const int bbase = ncol * KT_ + kk * 32 + kbB;
#pragma unroll
      for (int e = 0; e < 16; ++e) bfb[e] = Bs[wave][bbase + ((e < 8) ? 0 : 16) + (e & 7)];
      acc = __builtin_amdgcn_wmma_f32_16x16x32_bf16(false, af, false, bfb,
                                                    (short)0, acc, false, false);
    }
    __syncthreads();
  }

  // epilogue: C layout: VGPR r, lanes 0-15 -> m=r, lanes 16-31 -> m=8+r; n = lane%16
#pragma unroll
  for (int r = 0; r < 8; ++r) {
    int m = m0 + r + ((lane < 16) ? 0 : 8);
    int n = n0 + (lane & 15);
    if (m < M && n < N) {
      float v = acc[r];
      if (bias) v += bias[n];
      v = act_f(v, act);
      if (out_bf) ((bf16_t*)Cout)[(size_t)m * ldc + n] = (bf16_t)v;
      else        ((float*)Cout)[(size_t)m * ldc + n]  = v;
    }
  }
}

// ---------------- KAN operand packing ----------------
// Aext[r, 0:fin] = silu(x), Aext[r, fin + i*8 + k] = basis_k(x_i)
__global__ void kan_prep(const float* __restrict__ X, bf16_t* __restrict__ Aext,
                         int rows, int fin)
{
  long long idx = (long long)blockIdx.x * blockDim.x + threadIdx.x;
  if (idx >= (long long)rows * fin) return;
  int i = (int)(idx % fin);
  int r = (int)(idx / fin);
  float x = X[(size_t)r * fin + i];
  int lde = fin * 9;
  Aext[(size_t)r * lde + i] = (bf16_t)silu_f(x);
  float bas[8];
  bsplines(x, bas);
#pragma unroll
  for (int k = 0; k < 8; ++k)
    Aext[(size_t)r * lde + fin + i * 8 + k] = (bf16_t)bas[k];
}

__global__ void kan_pack_w(const float* __restrict__ bw, const float* __restrict__ sw,
                           bf16_t* __restrict__ Bext, int fout, int fin)
{
  long long idx = (long long)blockIdx.x * blockDim.x + threadIdx.x;
  if (idx >= (long long)fout * fin) return;
  int i = (int)(idx % fin);
  int o = (int)(idx / fin);
  int lde = fin * 9;
  Bext[(size_t)o * lde + i] = (bf16_t)bw[(size_t)o * fin + i];
#pragma unroll
  for (int k = 0; k < 8; ++k)
    Bext[(size_t)o * lde + fin + i * 8 + k] = (bf16_t)sw[((size_t)o * fin + i) * 8 + k];
}

// KAN with fin==1 (scalar in); optional dual bf16 output
__global__ void kan_f1(const float* __restrict__ X, float* __restrict__ Y,
                       bf16_t* __restrict__ Ybf,
                       const float* __restrict__ bw, const float* __restrict__ sw,
                       int rows, int fout)
{
  long long idx = (long long)blockIdx.x * blockDim.x + threadIdx.x;
  if (idx >= (long long)rows * fout) return;
  int o = (int)(idx % fout);
  int r = (int)(idx / fout);
  float x = X[r];
  float bas[8];
  bsplines(x, bas);
  float y = silu_f(x) * bw[o];
#pragma unroll
  for (int k = 0; k < 8; ++k) y += bas[k] * sw[o * 8 + k];
  Y[idx] = y;
  if (Ybf) Ybf[idx] = (bf16_t)y;
}

// KAN with fout==1: wave-per-row reduction
__global__ void kan_t1(const float* __restrict__ X, float* __restrict__ Y,
                       const float* __restrict__ bw, const float* __restrict__ sw,
                       int rows, int fin)
{
  int wid  = blockIdx.x * (blockDim.x >> 5) + (threadIdx.x >> 5);
  int lane = threadIdx.x & 31;
  if (wid >= rows) return;
  float s = 0.f;
  for (int i = lane; i < fin; i += 32) {
    float x = X[(size_t)wid * fin + i];
    s += silu_f(x) * bw[i];
    float bas[8];
    bsplines(x, bas);
#pragma unroll
    for (int k = 0; k < 8; ++k) s += bas[k] * sw[i * 8 + k];
  }
#pragma unroll
  for (int m = 16; m >= 1; m >>= 1) s += __shfl_xor(s, m, 32);
  if (lane == 0) Y[wid] = s;
}

// ---------------- add + RMS norm (wave per row, D<=512); optional bf16 dual out ----
__global__ void add_rms_k(const float* __restrict__ A, const float* __restrict__ Bb,
                          const float* __restrict__ w, float* __restrict__ Y,
                          bf16_t* __restrict__ Ybf, int rows, int D)
{
  int wid  = blockIdx.x * (blockDim.x >> 5) + (threadIdx.x >> 5);
  int lane = threadIdx.x & 31;
  if (wid >= rows) return;
  float vals[16];
  float ss = 0.f;
  int cnt = 0;
  for (int i = lane; i < D; i += 32) {
    float v = A[(size_t)wid * D + i];
    if (Bb) v += Bb[(size_t)wid * D + i];
    vals[cnt++] = v;
    ss += v * v;
  }
#pragma unroll
  for (int m = 16; m >= 1; m >>= 1) ss += __shfl_xor(ss, m, 32);
  float sc = rsqrtf(ss / (float)D + EPS_);
  cnt = 0;
  for (int i = lane; i < D; i += 32) {
    float v = vals[cnt++] * sc * w[i];
    Y[(size_t)wid * D + i] = v;
    if (Ybf) Ybf[(size_t)wid * D + i] = (bf16_t)v;
  }
}

// ---------------- Mamba pieces ----------------
// depthwise causal conv over sequence (DCONV=4) + bias + silu; xz rows x 2048, xi = cols[0:1024)
__global__ void conv_silu(const bf16_t* __restrict__ xz, const float* __restrict__ cw,
                          const float* __restrict__ cb, bf16_t* __restrict__ out,
                          int Lp, long long n)
{
  long long idx = (long long)blockIdx.x * blockDim.x + threadIdx.x;
  if (idx >= n) return;
  int d  = (int)(idx & 1023);
  long long bl = idx >> 10;
  int l = (int)(bl % Lp);
  long long b = bl / Lp;
  float s = cb[d];
#pragma unroll
  for (int kk = 0; kk < 4; ++kk) {
    int ls = l + kk - 3;
    if (ls >= 0) s += (float)xz[((b * Lp + ls) * 2048) + d] * cw[d * 4 + kk];
  }
  out[idx] = (bf16_t)silu_f(s);
}

// selective scan: block per batch, 1024 threads = 1024 channels (32 waves, wave32)
// fused epilogue: y = (scan + D*xi) * silu(z) -> bf16
__global__ void mamba_scan(const float* __restrict__ dt, const float* __restrict__ xdbl,
                           const bf16_t* __restrict__ convx, const bf16_t* __restrict__ xz,
                           const float* __restrict__ A_log, const float* __restrict__ Dp,
                           bf16_t* __restrict__ ybf, int Lp)
{
  __shared__ float bs[16], cs[16];
  const int b = blockIdx.x;
  const int d = threadIdx.x;
  float a[16], h[16];
#pragma unroll
  for (int n = 0; n < 16; ++n) {
    a[n] = -__expf(A_log[d * 16 + n]);
    h[n] = 0.f;
  }
  const float Dd = Dp[d];
  for (int t = 0; t < Lp; ++t) {
    size_t row = (size_t)b * Lp + t;
    if (threadIdx.x < 32) {
      float v = xdbl[row * 64 + 32 + threadIdx.x];
      if (threadIdx.x < 16) bs[threadIdx.x] = v;
      else                  cs[threadIdx.x - 16] = v;
    }
    __syncthreads();
    float dtv = dt[row * 1024 + d];
    float xv  = (float)convx[row * 1024 + d];
    float y = 0.f;
#pragma unroll
    for (int n = 0; n < 16; ++n) {
      h[n] = __expf(dtv * a[n]) * h[n] + dtv * xv * bs[n];
      y += h[n] * cs[n];
    }
    y += Dd * xv;
    float zv = (float)xz[row * 2048 + 1024 + d];
    ybf[row * 1024 + d] = (bf16_t)(y * silu_f(zv));
    __syncthreads();
  }
}

// ---------------- attention core (tiny; thread per (b,h,sq)) ----------------
__global__ void mha_attn(const float* __restrict__ qh, const float* __restrict__ kh,
                         const float* __restrict__ vh, float* __restrict__ o, int Sk)
{
  int idx = blockIdx.x * blockDim.x + threadIdx.x;
  if (idx >= B_ * HEADS_ * SEQ_) return;
  int sq = idx % SEQ_;
  int t  = idx / SEQ_;
  int hh = t % HEADS_;
  int b  = t / HEADS_;
  float q[16];
#pragma unroll
  for (int j = 0; j < 16; ++j) q[j] = qh[((b * SEQ_ + sq) * 64) + hh * 16 + j];
  float sc[SEQ_];
  float mx = -1e30f;
  for (int s = 0; s < Sk; ++s) {
    float d = 0.f;
#pragma unroll
    for (int j = 0; j < 16; ++j) d += q[j] * kh[((b * Sk + s) * 64) + hh * 16 + j];
    d *= 0.25f;  // 1/sqrt(16)
    sc[s] = d;
    mx = fmaxf(mx, d);
  }
  float sum = 0.f;
  for (int s = 0; s < Sk; ++s) { sc[s] = __expf(sc[s] - mx); sum += sc[s]; }
  float inv = 1.f / sum;
  float ov[16] = {};
  for (int s = 0; s < Sk; ++s) {
    float wgt = sc[s] * inv;
#pragma unroll
    for (int j = 0; j < 16; ++j) ov[j] += wgt * vh[((b * Sk + s) * 64) + hh * 16 + j];
  }
#pragma unroll
  for (int j = 0; j < 16; ++j) o[((b * SEQ_ + sq) * 64) + hh * 16 + j] = ov[j];
}

// ---------------- small glue kernels ----------------
__global__ void pack_bf(const float* __restrict__ X, bf16_t* __restrict__ Y, long long n)
{
  long long i = (long long)blockIdx.x * blockDim.x + threadIdx.x;
  if (i < n) Y[i] = (bf16_t)X[i];
}

__global__ void build_xs(const float* __restrict__ X, const float* __restrict__ st,
                         float* __restrict__ xs)
{
  int idx = blockIdx.x * blockDim.x + threadIdx.x;
  if (idx >= B_ * SEQ_ * 19) return;
  int c = idx % 19;
  int t = idx / 19;
  int s = t % SEQ_;
  int b = t / SEQ_;
  xs[idx] = (c < 16) ? X[(b * SEQ_ + s) * 17 + c] : st[b * 3 + (c - 16)];
}

__global__ void lc_gather(const float* __restrict__ X, const float* __restrict__ emb,
                          float* __restrict__ out)
{
  int idx = blockIdx.x * blockDim.x + threadIdx.x;
  if (idx >= B_ * SEQ_) return;
  int k = (int)X[idx * 17 + 16];
  k = min(max(k, 0), 16);
  out[idx] = emb[k];
}

__global__ void build_pos2(const float* __restrict__ lat, const float* __restrict__ lon,
                           float* __restrict__ p)
{
  int b = blockIdx.x * blockDim.x + threadIdx.x;
  if (b >= B_) return;
  p[b * 2 + 0] = lat[b];
  p[b * 2 + 1] = lon[b];
}

// ---------------- parameter leaf indices (jax tree order: sorted dict keys) ----------------
enum {
  IX_X = 0, IX_LAT, IX_LON, IX_ST,
  P_LC_EMB = 4, P_LCENC_BW = 5, P_LCENC_SW = 6,
  CA1 = 7, CA2 = 23,
  P_DYN_BW = 39, P_DYN_SW = 40, P_DND_BW = 41, P_DND_SW = 42,
  FM0 = 43, FM1 = 58, FM2 = 73, FM3 = 88, FMF = 112,
  P_PEK1_BW = 136, P_PEK1_SW = 137, P_PEK2_BW = 138, P_PEK2_SW = 139,
  P_PE_RMS = 140, P_PE_RMS2 = 141,
  P_REG1_BW = 142, P_REG1_SW = 143, P_REG2_BW = 144, P_REG2_SW = 145,
  P_REG3_BW = 146, P_REG3_SW = 147,
  P_SEK1_BW = 148, P_SEK1_SW = 149, P_SEK2_BW = 150, P_SEK2_SW = 151, P_SE_RMS = 152
};
// cross-attn leaf offsets (sorted keys)
enum { CA_BK = 0, CA_BO, CA_BQ, CA_BV, CA_FF1BW, CA_FF1SW, CA_FF2BW, CA_FF2SW,
       CA_FF3BW, CA_FF3SW, CA_LN1, CA_LN2, CA_WK, CA_WO, CA_WQ, CA_WV };
// mamba leaf offsets within layer (sorted): A_log,D,b_dt,conv_b,conv_w,w_dt,w_in,w_out,w_x
enum { M_ALOG = 0, M_D, M_BDT, M_CONVB, M_CONVW, M_WDT, M_WIN, M_WOUT, M_WX };

extern "C" void kernel_launch(void* const* d_in, const int* in_sizes, int n_in,
                              void* d_out, int out_size, void* d_ws, size_t ws_size,
                              hipStream_t stream)
{
  (void)in_sizes; (void)n_in; (void)out_size; (void)ws_size;
  hipStream_t st = stream;
  auto F = [&](int i) -> const float* { return (const float*)d_in[i]; };

  // bump allocator over workspace
  char* base = (char*)d_ws;
  size_t off = 0;
  auto alloc = [&](size_t bytes) -> void* {
    size_t a = (off + 255) & ~(size_t)255;
    off = a + bytes;
    return (void*)(base + a);
  };

  const int R2MAX = B_ * SEQ_ * EMBED_;  // 5120 max inner-mamba rows
  // mamba-stage scratch (reused across all feature_mamba calls)
  float*  t0    = (float*)alloc((size_t)R2MAX * 512 * 4);
  bf16_t* t0b   = (bf16_t*)alloc((size_t)R2MAX * 512 * 2);
  bf16_t* xzb   = (bf16_t*)alloc((size_t)R2MAX * 2048 * 2);
  bf16_t* convb = (bf16_t*)alloc((size_t)R2MAX * 1024 * 2);
  float*  xdbl  = (float*)alloc((size_t)R2MAX * 64 * 4);
  bf16_t* xdblb = (bf16_t*)alloc((size_t)R2MAX * 64 * 2);
  float*  dtb   = (float*)alloc((size_t)R2MAX * 1024 * 4);
  bf16_t* yb    = (bf16_t*)alloc((size_t)R2MAX * 1024 * 2);
  float*  mout  = (float*)alloc((size_t)R2MAX * 512 * 4);
  float*  tv    = (float*)alloc((size_t)R2MAX * 4);
  bf16_t* w_in_b  = (bf16_t*)alloc((size_t)2048 * 512 * 2);
  bf16_t* w_x_b   = (bf16_t*)alloc((size_t)64 * 1024 * 2);
  bf16_t* w_dt_b  = (bf16_t*)alloc((size_t)1024 * 32 * 2);
  bf16_t* w_out_b = (bf16_t*)alloc((size_t)512 * 1024 * 2);
  // small activations
  float* xsb  = (float*)alloc(80 * 19 * 4);
  float* lcx  = (float*)alloc(80 * 4);
  float* lcE  = (float*)alloc(80 * 64 * 4);
  float* pos2 = (float*)alloc(8 * 2 * 4);
  float* h128 = (float*)alloc(80 * 128 * 4);
  float* posE = (float*)alloc(8 * 64 * 4);
  float* se3  = (float*)alloc(8 * 3 * 4);
  float* seE  = (float*)alloc(8 * 64 * 4);
  float* dbuf = (float*)alloc(80 * 64 * 4);
  float* ddb  = (float*)alloc(80 * 20 * 4);
  float* r20  = (float*)alloc(80 * 20 * 4);
  // attention scratch
  bf16_t* qpb = (bf16_t*)alloc(80 * 64 * 2);
  bf16_t* kvb = (bf16_t*)alloc(80 * 64 * 2);
  bf16_t* opb = (bf16_t*)alloc(80 * 64 * 2);
  bf16_t* wqb = (bf16_t*)alloc(4096 * 2);
  bf16_t* wkb = (bf16_t*)alloc(4096 * 2);
  bf16_t* wvb = (bf16_t*)alloc(4096 * 2);
  bf16_t* wob = (bf16_t*)alloc(4096 * 2);
  float* qhB = (float*)alloc(80 * 64 * 4);
  float* khB = (float*)alloc(80 * 64 * 4);
  float* vhB = (float*)alloc(80 * 64 * 4);
  float* ohB = (float*)alloc(80 * 64 * 4);
  float* atn = (float*)alloc(80 * 64 * 4);
  float* o1  = (float*)alloc(80 * 64 * 4);
  float* h2b = (float*)alloc(80 * 64 * 4);
  float* h3b = (float*)alloc(80 * 64 * 4);
  // KAN operand scratch
  bf16_t* aext = (bf16_t*)alloc((size_t)80 * 1152 * 2);
  bf16_t* bext = (bf16_t*)alloc((size_t)128 * 1152 * 2);

  // ---- host-side helpers ----
  auto gemm = [&](const bf16_t* A, int lda, const bf16_t* Bw, int ldb, const float* bias,
                  void* C, int ldc, int outbf, int M, int N, int K, int act) {
    dim3 g((unsigned)((N + 63) / 64), (unsigned)((M + 15) / 16));
    gemm_bf16_wmma<<<g, 128, 0, st>>>(A, lda, Bw, ldb, bias, C, ldc, outbf, M, N, K, act);
  };
  auto packb = [&](const float* X, bf16_t* Y, long long n) {
    pack_bf<<<(unsigned)((n + 255) / 256), 256, 0, st>>>(X, Y, n);
  };
  auto addrms = [&](const float* A, const float* Bb, const float* w, float* Y,
                    bf16_t* Ybf, int rows, int D) {
    add_rms_k<<<(unsigned)((rows + 3) / 4), 128, 0, st>>>(A, Bb, w, Y, Ybf, rows, D);
  };
  auto kan_gen = [&](const float* X, int rows, int fin, int fout, int bwI, int swI,
                     int act, float* Y) {
    long long n1 = (long long)rows * fin;
    kan_prep<<<(unsigned)((n1 + 255) / 256), 256, 0, st>>>(X, aext, rows, fin);
    long long n2 = (long long)fout * fin;
    kan_pack_w<<<(unsigned)((n2 + 255) / 256), 256, 0, st>>>(F(bwI), F(swI), bext, fout, fin);
    gemm(aext, fin * 9, bext, fin * 9, nullptr, Y, fout, 0, rows, fout, fin * 9, act);
  };
  auto feature_mamba = [&](float* x, int L, int D, int fmBase, int nl) {
    int rows2 = B_ * L * D;
    int Lp = L * D;
    int te = fmBase + 9 * nl;  // ln, ln_x, tmpembed.bw/sw, tmpembed_t.bw/sw
    long long ne = (long long)rows2 * 512;
    kan_f1<<<(unsigned)((ne + 255) / 256), 256, 0, st>>>(x, t0, t0b,
                                                         F(te + 2), F(te + 3), rows2, 512);
    for (int l = 0; l < nl; ++l) {
      int mb = fmBase + 9 * l;
      packb(F(mb + M_WIN),  w_in_b,  (long long)2048 * 512);
      packb(F(mb + M_WX),   w_x_b,   (long long)64 * 1024);
      packb(F(mb + M_WDT),  w_dt_b,  (long long)1024 * 32);
      packb(F(mb + M_WOUT), w_out_b, (long long)512 * 1024);
      gemm(t0b, 512, w_in_b, 512, nullptr, xzb, 2048, 1, rows2, 2048, 512, 0);
      long long nc = (long long)rows2 * 1024;
      conv_silu<<<(unsigned)((nc + 255) / 256), 256, 0, st>>>(xzb, F(mb + M_CONVW),
                                                              F(mb + M_CONVB), convb, Lp, nc);
      gemm(convb, 1024, w_x_b, 1024, nullptr, xdbl, 64, 0, rows2, 64, 1024, 0);
      packb(xdbl, xdblb, (long long)rows2 * 64);
      gemm(xdblb, 64, w_dt_b, 32, F(mb + M_BDT), dtb, 1024, 0, rows2, 1024, 32, 3); // softplus
      mamba_scan<<<B_, 1024, 0, st>>>(dtb, xdbl, convb, xzb, F(mb + M_ALOG), F(mb + M_D), yb, Lp);
      gemm(yb, 1024, w_out_b, 1024, nullptr, mout, 512, 0, rows2, 512, 1024, 0);
      addrms(mout, t0, F(te + 0), t0, t0b, rows2, 512);  // residual + rms, dual f32/bf16
    }
    kan_t1<<<(unsigned)((rows2 + 7) / 8), 256, 0, st>>>(t0, tv, F(te + 4), F(te + 5), rows2, 512);
    addrms(tv, x, F(te + 1), x, nullptr, B_ * L, D);
  };
  auto cross_attn = [&](float* q, const float* kv, int Sk, int ca) {
    int rk = B_ * Sk;
    packb(q, qpb, 80 * 64);
    packb(kv, kvb, (long long)rk * 64);
    packb(F(ca + CA_WQ), wqb, 4096);
    packb(F(ca + CA_WK), wkb, 4096);
    packb(F(ca + CA_WV), wvb, 4096);
    packb(F(ca + CA_WO), wob, 4096);
    gemm(qpb, 64, wqb, 64, F(ca + CA_BQ), qhB, 64, 0, 80, 64, 64, 0);
    gemm(kvb, 64, wkb, 64, F(ca + CA_BK), khB, 64, 0, rk, 64, 64, 0);
    gemm(kvb, 64, wvb, 64, F(ca + CA_BV), vhB, 64, 0, rk, 64, 64, 0);
    mha_attn<<<(B_ * HEADS_ * SEQ_ + 63) / 64, 64, 0, st>>>(qhB, khB, vhB, ohB, Sk);
    packb(ohB, opb, 80 * 64);
    gemm(opb, 64, wob, 64, F(ca + CA_BO), atn, 64, 0, 80, 64, 64, 0);
    addrms(q, atn, F(ca + CA_LN1), o1, nullptr, 80, 64);
    kan_gen(o1, 80, 64, 128, ca + CA_FF1BW, ca + CA_FF1SW, 2, h128);   // gelu
    kan_gen(h128, 80, 128, 64, ca + CA_FF2BW, ca + CA_FF2SW, 2, h2b);  // gelu
    kan_gen(h2b, 80, 64, 64, ca + CA_FF3BW, ca + CA_FF3SW, 2, h3b);    // gelu
    addrms(o1, h3b, F(ca + CA_LN2), q, nullptr, 80, 64);
  };

  // ---- forward pass ----
  build_xs<<<(B_ * SEQ_ * 19 + 255) / 256, 256, 0, st>>>(F(IX_X), F(IX_ST), xsb);
  lc_gather<<<1, 128, 0, st>>>(F(IX_X), F(P_LC_EMB), lcx);
  kan_f1<<<(80 * 64 + 255) / 256, 256, 0, st>>>(lcx, lcE, nullptr,
                                                F(P_LCENC_BW), F(P_LCENC_SW), 80, 64);

  build_pos2<<<1, 32, 0, st>>>(F(IX_LAT), F(IX_LON), pos2);
  addrms(pos2, nullptr, F(P_PE_RMS2), pos2, nullptr, 8, 2);
  kan_gen(pos2, 8, 2, 128, P_PEK1_BW, P_PEK1_SW, 0, h128);
  kan_gen(h128, 8, 128, 64, P_PEK2_BW, P_PEK2_SW, 0, posE);
  addrms(posE, nullptr, F(P_PE_RMS), posE, nullptr, 8, 64);

  addrms(F(IX_ST), nullptr, F(P_SE_RMS), se3, nullptr, 8, 3);
  kan_gen(se3, 8, 3, 128, P_SEK1_BW, P_SEK1_SW, 0, h128);
  kan_gen(h128, 8, 128, 64, P_SEK2_BW, P_SEK2_SW, 0, seE);

  feature_mamba(xsb, SEQ_, 19, FM0, 1);
  kan_gen(xsb, 80, 19, 64, P_DYN_BW, P_DYN_SW, 0, dbuf);

  cross_attn(dbuf, lcE, SEQ_, CA1);
  feature_mamba(dbuf, SEQ_, 64, FM1, 1);
  cross_attn(dbuf, seE, 1, CA2);
  feature_mamba(dbuf, SEQ_, 64, FM2, 1);
  cross_attn(dbuf, posE, 1, CA1);
  feature_mamba(dbuf, SEQ_, 64, FM3, 2);

  kan_gen(dbuf, 80, 64, 20, P_DND_BW, P_DND_SW, 0, ddb);
  feature_mamba(ddb, SEQ_, 20, FMF, 2);

  kan_gen(ddb, 80, 20, 128, P_REG1_BW, P_REG1_SW, 0, h128);
  kan_gen(h128, 80, 128, 20, P_REG2_BW, P_REG2_SW, 0, r20);
  kan_gen(r20, 80, 20, 1, P_REG3_BW, P_REG3_SW, 0, (float*)d_out);
}
